// SiameseGNN_GAT_78116865180411
// MI455X (gfx1250) — compile-verified, matrix operand order checked by reading, not run on hardware
//
#include <hip/hip_runtime.h>
#include <hip/hip_bf16.h>
#include <math.h>

// ---------------------------------------------------------------------------
// Problem constants (from reference)
// ---------------------------------------------------------------------------
#define NN    50000
#define EE    800000
#define MM    (EE + NN)     // edges + self loops
#define DD    128           // feature dim for both conv layers (D_IN = H*NH = 128)
#define NHH   64            // per-head channels
#define TOPK  512
#define PADN  65536         // next pow2 >= NN for bitonic sort

typedef __attribute__((ext_vector_type(16))) __bf16 v16bf;
typedef __attribute__((ext_vector_type(8)))  float  v8f;

// float -> bf16 with round-to-nearest-even (pure integer ops, always compiles)
__device__ __forceinline__ __bf16 f2bf(float f) {
  unsigned u = __float_as_uint(f);
  unsigned r = (u + 0x7fffu + ((u >> 16) & 1u)) >> 16;
  unsigned short hs = (unsigned short)r;
  __bf16 b;
  __builtin_memcpy(&b, &hs, 2);
  return b;
}

// sortable-uint encode/decode for float atomic max
__device__ __forceinline__ unsigned fenc(float f) {
  unsigned u = __float_as_uint(f);
  return (u & 0x80000000u) ? ~u : (u | 0x80000000u);
}
__device__ __forceinline__ float fdec(unsigned u) {
  unsigned b = (u & 0x80000000u) ? (u ^ 0x80000000u) : ~u;
  return __uint_as_float(b);
}

// ---------------------------------------------------------------------------
// Utility fills / converts
// ---------------------------------------------------------------------------
__global__ void k_fill_f32(float* p, float v, size_t n) {
  size_t i = (size_t)blockIdx.x * blockDim.x + threadIdx.x;
  if (i < n) p[i] = v;
}
__global__ void k_fill_u32(unsigned* p, unsigned v, size_t n) {
  size_t i = (size_t)blockIdx.x * blockDim.x + threadIdx.x;
  if (i < n) p[i] = v;
}
__global__ void k_f32_to_bf16(const float* __restrict__ in, __bf16* __restrict__ out, size_t n) {
  size_t i = (size_t)blockIdx.x * blockDim.x + threadIdx.x;
  if (i < n) out[i] = f2bf(in[i]);
}
// Wt[n][k] = W[k][n]  (W row-major [128,128]) -> bf16, so B-fragment loads are contiguous
__global__ void k_transpose_bf16(const float* __restrict__ W, __bf16* __restrict__ Wt) {
  int t = blockIdx.x * blockDim.x + threadIdx.x;
  if (t >= DD * DD) return;
  int n = t >> 7, k = t & 127;
  Wt[t] = f2bf(W[k * DD + n]);
}

// ---------------------------------------------------------------------------
// GEMM: C[N,128] = A[N,128](bf16) * W[128,128]  via v_wmma_f32_16x16x32_bf16
// One wave per 16-row strip; 4 k-steps held in registers; loop 8 column tiles.
// ---------------------------------------------------------------------------
__global__ __launch_bounds__(32) void k_gemm_bf16(
    const __bf16* __restrict__ A, const __bf16* __restrict__ Bt,
    float* __restrict__ C) {
  const int m0     = blockIdx.x * 16;
  const int lane   = threadIdx.x & 31;
  const int lane15 = lane & 15;
  const int khalf  = (lane >> 4) * 8;   // lanes 16-31 take K+8 / K+24 halves

  v16bf a[4];
  const __bf16* pa = A + (size_t)(m0 + lane15) * DD;
#pragma unroll
  for (int ks = 0; ks < 4; ++ks) {
    const int kb = ks * 32 + khalf;
#pragma unroll
    for (int i = 0; i < 8; ++i) {
      a[ks][i]     = pa[kb + i];        // K = kb .. kb+7
      a[ks][8 + i] = pa[kb + 16 + i];   // K = kb+16 .. kb+23
    }
  }

#pragma unroll
  for (int nt = 0; nt < 8; ++nt) {
    const int n0 = nt * 16;
    const __bf16* pb = Bt + (size_t)(n0 + lane15) * DD;  // column n0+lane15 of W
    v8f acc = {};
#pragma unroll
    for (int ks = 0; ks < 4; ++ks) {
      const int kb = ks * 32 + khalf;
      v16bf b;
#pragma unroll
      for (int i = 0; i < 8; ++i) {
        b[i]     = pb[kb + i];
        b[8 + i] = pb[kb + 16 + i];
      }
      acc = __builtin_amdgcn_wmma_f32_16x16x32_bf16(
          false, a[ks], false, b, (short)0, acc, false, false);
    }
    // C/D layout: VGPR r -> row m0 + r + khalf ; col n0 + lane15
    float* pc = C + (size_t)(m0 + khalf) * DD + n0 + lane15;
#pragma unroll
    for (int r = 0; r < 8; ++r) pc[(size_t)r * DD] = acc[r];
  }
}

// ---------------------------------------------------------------------------
// GAT attention pipeline
// ---------------------------------------------------------------------------
// al_s[n,h] = sum_c xp[n,h,c]*a_src[h,c]  (and same for a_dst)
__global__ void k_node_attn(const float* __restrict__ xp,
                            const float* __restrict__ a_src,
                            const float* __restrict__ a_dst,
                            float* __restrict__ als, float* __restrict__ ald) {
  int t = blockIdx.x * blockDim.x + threadIdx.x;
  if (t >= NN * 2) return;
  int node = t >> 1, h = t & 1;
  const float* xr  = xp + (size_t)node * DD + h * NHH;
  const float* as_ = a_src + h * NHH;
  const float* ad_ = a_dst + h * NHH;
  float s = 0.f, d = 0.f;
#pragma unroll 8
  for (int c = 0; c < NHH; ++c) { float v = xr[c]; s += v * as_[c]; d += v * ad_[c]; }
  als[t] = s; ald[t] = d;
}

// pass1: e = leaky_relu(al_s[src]+al_d[dst], 0.2); segment max over dst
__global__ void k_edge_pass1(const int* __restrict__ ei,
                             const float* __restrict__ als, const float* __restrict__ ald,
                             float* __restrict__ ebuf, unsigned* __restrict__ maxu) {
  int m = blockIdx.x * blockDim.x + threadIdx.x;
  if (m >= MM) return;
  int s, d;
  if (m < EE) { s = ei[m]; d = ei[EE + m]; } else { s = d = m - EE; }
#pragma unroll
  for (int h = 0; h < 2; ++h) {
    float e = als[s * 2 + h] + ald[d * 2 + h];
    e = e > 0.f ? e : 0.2f * e;
    ebuf[(size_t)m * 2 + h] = e;
    atomicMax(&maxu[d * 2 + h], fenc(e));
  }
}

// pass2: ex = exp(e - max[dst]); segment sum (denom pre-filled with 1e-16)
__global__ void k_edge_pass2(const int* __restrict__ ei,
                             float* __restrict__ ebuf,
                             const unsigned* __restrict__ maxu,
                             float* __restrict__ denom) {
  int m = blockIdx.x * blockDim.x + threadIdx.x;
  if (m >= MM) return;
  int d;
  if (m < EE) { d = ei[EE + m]; } else { d = m - EE; }
#pragma unroll
  for (int h = 0; h < 2; ++h) {
    size_t idx = (size_t)m * 2 + h;
    float ex = expf(ebuf[idx] - fdec(maxu[d * 2 + h]));
    ebuf[idx] = ex;
    atomicAdd(&denom[d * 2 + h], ex);
  }
}

// pass3: out[dst] += (ex/denom[dst]) * xp[src]   -- one wave32 per edge, float4/lane
__global__ __launch_bounds__(256) void k_edge_scatter(
    const int* __restrict__ ei, const float* __restrict__ ebuf,
    const float* __restrict__ denom, const float* __restrict__ xp,
    float* __restrict__ acc) {
  int w = blockIdx.x * 8 + (threadIdx.x >> 5);
  if (w >= MM) return;
  int lane = threadIdx.x & 31;
  int s, d;
  if (w < EE) { s = ei[w]; d = ei[EE + w]; } else { s = d = w - EE; }
  int c0 = lane * 4;
  int h  = c0 >> 6;  // channels 0..63 -> head 0, 64..127 -> head 1
  float alpha = ebuf[(size_t)w * 2 + h] / denom[d * 2 + h];
  const float4 v = *(const float4*)(xp + (size_t)s * DD + c0);
  float* o = acc + (size_t)d * DD + c0;
  atomicAdd(o + 0, alpha * v.x);
  atomicAdd(o + 1, alpha * v.y);
  atomicAdd(o + 2, alpha * v.z);
  atomicAdd(o + 3, alpha * v.w);
}

__global__ void k_bias_relu_bf16(const float* __restrict__ acc, const float* __restrict__ bias,
                                 __bf16* __restrict__ out, size_t n) {
  size_t i = (size_t)blockIdx.x * blockDim.x + threadIdx.x;
  if (i >= n) return;
  float v = acc[i] + bias[i & (DD - 1)];
  out[i] = f2bf(v > 0.f ? v : 0.f);
}
__global__ void k_bias_relu_f32(const float* __restrict__ acc, const float* __restrict__ bias,
                                float* __restrict__ out, size_t n) {
  size_t i = (size_t)blockIdx.x * blockDim.x + threadIdx.x;
  if (i >= n) return;
  float v = acc[i] + bias[i & (DD - 1)];
  out[i] = v > 0.f ? v : 0.f;
}

// ---------------------------------------------------------------------------
// Pairwise distance + bitonic top-k
// ---------------------------------------------------------------------------
__global__ void k_pairdist(const float* __restrict__ o1, const float* __restrict__ o2,
                           float* __restrict__ simpad) {
  int t = blockIdx.x * blockDim.x + threadIdx.x;
  if (t >= PADN) return;
  if (t >= NN) { simpad[t] = -INFINITY; return; }
  const float* p1 = o1 + (size_t)t * DD;
  const float* p2 = o2 + (size_t)t * DD;
  float ss = 0.f;
#pragma unroll 8
  for (int c = 0; c < DD; ++c) { float df = p1[c] - p2[c] + 1e-6f; ss += df * df; }
  simpad[t] = sqrtf(ss);
}

// one compare-exchange step of a descending bitonic sort
__global__ void k_bitonic(float* __restrict__ d, int j, int k) {
  int i = blockIdx.x * blockDim.x + threadIdx.x;
  int ixj = i ^ j;
  if (ixj > i) {
    float a = d[i], b = d[ixj];
    bool sw = ((i & k) == 0) ? (a < b) : (a > b);
    if (sw) { d[i] = b; d[ixj] = a; }
  }
}

// ---------------------------------------------------------------------------
// MLP head (tiny): fc+LN+relu blocks with LDS reductions
// ---------------------------------------------------------------------------
__global__ __launch_bounds__(128) void k_fc1(const float* __restrict__ topv,
    const float* __restrict__ w, const float* __restrict__ b,
    const float* __restrict__ lnw, const float* __restrict__ lnb,
    float* __restrict__ h1) {
  int j = threadIdx.x;  // 0..127
  float acc = b[j];
  for (int k = 0; k < TOPK; ++k) acc += topv[k] * w[k * 128 + j];
  __shared__ float red[128];
  red[j] = acc; __syncthreads();
  for (int s = 64; s > 0; s >>= 1) { if (j < s) red[j] += red[j + s]; __syncthreads(); }
  float mu = red[0] / 128.f; __syncthreads();
  float dv = acc - mu;
  red[j] = dv * dv; __syncthreads();
  for (int s = 64; s > 0; s >>= 1) { if (j < s) red[j] += red[j + s]; __syncthreads(); }
  float var = red[0] / 128.f;
  float y = dv * rsqrtf(var + 1e-5f) * lnw[j] + lnb[j];
  h1[j] = y > 0.f ? y : 0.f;
}

__global__ __launch_bounds__(64) void k_fc2(const float* __restrict__ h1,
    const float* __restrict__ w, const float* __restrict__ b,
    const float* __restrict__ lnw, const float* __restrict__ lnb,
    float* __restrict__ h2) {
  int j = threadIdx.x;  // 0..63
  float acc = b[j];
  for (int k = 0; k < 128; ++k) acc += h1[k] * w[k * 64 + j];
  __shared__ float red[64];
  red[j] = acc; __syncthreads();
  for (int s = 32; s > 0; s >>= 1) { if (j < s) red[j] += red[j + s]; __syncthreads(); }
  float mu = red[0] / 64.f; __syncthreads();
  float dv = acc - mu;
  red[j] = dv * dv; __syncthreads();
  for (int s = 32; s > 0; s >>= 1) { if (j < s) red[j] += red[j + s]; __syncthreads(); }
  float var = red[0] / 64.f;
  float y = dv * rsqrtf(var + 1e-5f) * lnw[j] + lnb[j];
  h2[j] = y > 0.f ? y : 0.f;
}

__global__ void k_fc3(const float* __restrict__ h2, const float* __restrict__ w,
                      const float* __restrict__ b, float* __restrict__ out) {
  if (threadIdx.x == 0 && blockIdx.x == 0) {
    float s = b[0];
    for (int k = 0; k < 64; ++k) s += h2[k] * w[k];
    out[0] = 1.f / (1.f + expf(-s));
  }
}

// ---------------------------------------------------------------------------
// Host orchestration
// ---------------------------------------------------------------------------
extern "C" void kernel_launch(void* const* d_in, const int* in_sizes, int n_in,
                              void* d_out, int out_size, void* d_ws, size_t ws_size,
                              hipStream_t stream) {
  (void)in_sizes; (void)n_in; (void)out_size; (void)ws_size;

  // workspace carve (256B aligned)
  char* p = (char*)d_ws;
  auto carve = [&](size_t bytes) -> void* {
    void* q = (void*)p;
    p += (bytes + 255) & ~(size_t)255;
    return q;
  };
  float*    o1     = (float*)   carve((size_t)NN * DD * 4);
  float*    o2     = (float*)   carve((size_t)NN * DD * 4);
  __bf16*   xb     = (__bf16*)  carve((size_t)NN * DD * 2);
  float*    xp     = (float*)   carve((size_t)NN * DD * 4);
  float*    acc    = (float*)   carve((size_t)NN * DD * 4);
  float*    ebuf   = (float*)   carve((size_t)MM * 2 * 4);
  float*    als    = (float*)   carve((size_t)NN * 2 * 4);
  float*    ald    = (float*)   carve((size_t)NN * 2 * 4);
  unsigned* maxu   = (unsigned*)carve((size_t)NN * 2 * 4);
  float*    denom  = (float*)   carve((size_t)NN * 2 * 4);
  __bf16*   W1t    = (__bf16*)  carve((size_t)DD * DD * 2);
  __bf16*   W2t    = (__bf16*)  carve((size_t)DD * DD * 2);
  float*    simpad = (float*)   carve((size_t)PADN * 4);
  float*    h1     = (float*)   carve(128 * 4);
  float*    h2     = (float*)   carve(64 * 4);

  const size_t nfeat = (size_t)NN * DD;
  const int gFeat  = (int)((nfeat + 255) / 256);
  const int gN2    = (NN * 2 + 255) / 256;
  const int gEdge  = (MM + 255) / 256;
  const int gScat  = (MM + 7) / 8;

  // pre-transpose weights to bf16 (B^T for contiguous WMMA B-fragment loads)
  k_transpose_bf16<<<(DD * DD + 255) / 256, 256, 0, stream>>>((const float*)d_in[4], W1t);
  k_transpose_bf16<<<(DD * DD + 255) / 256, 256, 0, stream>>>((const float*)d_in[8], W2t);

  for (int g = 0; g < 2; ++g) {
    const float* x  = (const float*)d_in[g];
    const int*   ei = (const int*)d_in[2 + g];
    float* outg = g ? o2 : o1;

    k_f32_to_bf16<<<gFeat, 256, 0, stream>>>(x, xb, nfeat);

    for (int layer = 0; layer < 2; ++layer) {
      const __bf16* Wt  = layer ? W2t : W1t;
      const float*  as_ = (const float*)d_in[layer ? 9  : 5];
      const float*  ad_ = (const float*)d_in[layer ? 10 : 6];
      const float*  bs  = (const float*)d_in[layer ? 11 : 7];

      k_gemm_bf16<<<NN / 16, 32, 0, stream>>>(xb, Wt, xp);
      k_node_attn<<<gN2, 256, 0, stream>>>(xp, as_, ad_, als, ald);
      k_fill_u32<<<gN2, 256, 0, stream>>>(maxu, 0u, (size_t)NN * 2);
      k_fill_f32<<<gN2, 256, 0, stream>>>(denom, 1e-16f, (size_t)NN * 2);
      k_fill_f32<<<gFeat, 256, 0, stream>>>(acc, 0.f, nfeat);
      k_edge_pass1<<<gEdge, 256, 0, stream>>>(ei, als, ald, ebuf, maxu);
      k_edge_pass2<<<gEdge, 256, 0, stream>>>(ei, ebuf, maxu, denom);
      k_edge_scatter<<<gScat, 256, 0, stream>>>(ei, ebuf, denom, xp, acc);
      if (layer == 0)
        k_bias_relu_bf16<<<gFeat, 256, 0, stream>>>(acc, bs, xb, nfeat);
      else
        k_bias_relu_f32<<<gFeat, 256, 0, stream>>>(acc, bs, outg, nfeat);
    }
  }

  // pairwise distance + pad, then full descending bitonic sort (top-512 sorted)
  k_pairdist<<<PADN / 256, 256, 0, stream>>>(o1, o2, simpad);
  for (int k = 2; k <= PADN; k <<= 1)
    for (int j = k >> 1; j > 0; j >>= 1)
      k_bitonic<<<PADN / 256, 256, 0, stream>>>(simpad, j, k);

  // MLP head; simpad[0..511] are the sorted top-k values
  k_fc1<<<1, 128, 0, stream>>>(simpad, (const float*)d_in[12], (const float*)d_in[13],
                               (const float*)d_in[14], (const float*)d_in[15], h1);
  k_fc2<<<1, 64, 0, stream>>>(h1, (const float*)d_in[16], (const float*)d_in[17],
                              (const float*)d_in[18], (const float*)d_in[19], h2);
  k_fc3<<<1, 32, 0, stream>>>(h2, (const float*)d_in[20], (const float*)d_in[21],
                              (float*)d_out);
}
